// SelfAttention_91182155694380
// MI455X (gfx1250) — compile-verified, matrix operand order checked by reading, not run on hardware
//
#include <hip/hip_runtime.h>

// ---------------------------------------------------------------------------
// Flash-attention for B=4, S=4096, D_in=128, D_out=64 on gfx1250 (wave32).
// Kernel 1: Q/K/V projections. Weights staged once per block into LDS
//           (transposed, f16) so B-fragments are contiguous ds_load_b128.
// Kernel 2: flash attention. 4 waves/block share one batch; K/V tiles are
//           double-buffered in LDS via async global->LDS copies (ASYNCcnt),
//           overlapping the next tile's copy with this tile's 8 WMMAs.
// ---------------------------------------------------------------------------

typedef __attribute__((ext_vector_type(16))) _Float16 v16h;
typedef __attribute__((ext_vector_type(8)))  float    v8f;
typedef __attribute__((__vector_size__(4 * sizeof(int)))) int v4i;

#define WMMA_F16(A, Bm, C) \
    __builtin_amdgcn_wmma_f32_16x16x32_f16(false, (A), false, (Bm), (short)0, (C), false, false)

constexpr int BATCH = 4;
constexpr int SEQ   = 4096;
constexpr int DIN   = 128;
constexpr int DOUT  = 64;
constexpr int BS    = BATCH * SEQ;
// fold 1/sqrt(64) and log2(e) into Q so softmax uses raw v_exp_f32 (exp2)
constexpr float QSCALE = 0.125f * 1.4426950408889634f;

static __device__ __forceinline__ float fast_exp2(float x) {
    return __builtin_amdgcn_exp2f(x);
}

// ---- async global->LDS copy (CDNA5), with portable fallback ---------------
#if defined(__has_builtin)
#if __has_builtin(__builtin_amdgcn_global_load_async_to_lds_b128) && \
    __has_builtin(__builtin_amdgcn_s_wait_asynccnt)
#define HAVE_ASYNC_LDS 1
#endif
#endif
#ifndef HAVE_ASYNC_LDS
#define HAVE_ASYNC_LDS 0
#endif

typedef __attribute__((address_space(1))) v4i g_v4i;   // global int4
typedef __attribute__((address_space(3))) v4i l_v4i;   // LDS int4

static __device__ __forceinline__ void async_copy16(const void* g, void* l) {
#if HAVE_ASYNC_LDS
    __builtin_amdgcn_global_load_async_to_lds_b128(
        (g_v4i*)(__UINTPTR_TYPE__)g, (l_v4i*)(__UINTPTR_TYPE__)l, 0, 0);
#else
    *(float4*)l = *(const float4*)g;   // global_load_b128 + ds_store_b128
#endif
}

template <int N>
static __device__ __forceinline__ void async_wait() {
#if HAVE_ASYNC_LDS
    __builtin_amdgcn_s_wait_asynccnt(N);
#endif
}

// ---------------------------------------------------------------------------
// Kernel 1: projections. All 4 waves of a block share projection p; W is
// staged to LDS transposed [n][k] as f16 once, then 16 WMMAs per wave.
// ---------------------------------------------------------------------------
__global__ __launch_bounds__(128) void proj_kernel(
        const float* __restrict__ x, const float* __restrict__ w,
        _Float16* __restrict__ Qo, _Float16* __restrict__ Ko,
        _Float16* __restrict__ Vt)
{
    __shared__ __attribute__((aligned(16))) _Float16 wlds[DOUT][DIN];  // 16 KB
    const int tid  = threadIdx.x;
    const int lane = tid & 31;
    const int warp = tid >> 5;
    const int gw   = blockIdx.x * 4 + warp;      // 0..3071
    const int p    = gw / (BS / 16);             // 0=Q 1=K 2=V (same whole block)
    const int m0   = (gw % (BS / 16)) * 16;
    const int hid  = lane >> 4;
    const int lr   = lane & 15;

    // stage weights: coalesced global reads, transposed f16 LDS writes
    const float* W = w + p * DIN * DOUT;
    #pragma unroll 8
    for (int i = 0; i < 64; ++i) {
        const int idx = i * 128 + tid;           // coalesced over tid
        wlds[idx & 63][idx >> 6] = (_Float16)W[idx];
    }
    __syncthreads();

    v8f c[4] = {v8f{}, v8f{}, v8f{}, v8f{}};

    for (int kc = 0; kc < DIN / 32; ++kc) {
        // A fragment from x: lane<16 -> row m0+lr, K {0..7,16..23}; lane>=16 +8
        const float* xr = x + (size_t)(m0 + lr) * DIN + kc * 32 + hid * 8;
        const float4 f0 = *(const float4*)(xr);
        const float4 f1 = *(const float4*)(xr + 4);
        const float4 f2 = *(const float4*)(xr + 16);
        const float4 f3 = *(const float4*)(xr + 20);
        v16h a;
        a[0]  = (_Float16)f0.x; a[1]  = (_Float16)f0.y; a[2]  = (_Float16)f0.z; a[3]  = (_Float16)f0.w;
        a[4]  = (_Float16)f1.x; a[5]  = (_Float16)f1.y; a[6]  = (_Float16)f1.z; a[7]  = (_Float16)f1.w;
        a[8]  = (_Float16)f2.x; a[9]  = (_Float16)f2.y; a[10] = (_Float16)f2.z; a[11] = (_Float16)f2.w;
        a[12] = (_Float16)f3.x; a[13] = (_Float16)f3.y; a[14] = (_Float16)f3.z; a[15] = (_Float16)f3.w;

        #pragma unroll
        for (int nt = 0; nt < 4; ++nt) {
            // B fragment from LDS: contiguous 16B runs -> ds_load_b128
            const _Float16* wc = &wlds[nt * 16 + lr][kc * 32 + hid * 8];
            v16h bf;
            #pragma unroll
            for (int j = 0; j < 8; ++j) { bf[j] = wc[j]; bf[8 + j] = wc[16 + j]; }
            c[nt] = WMMA_F16(a, bf, c[nt]);
        }
    }

    const float scale = (p == 0) ? QSCALE : 1.0f;
    if (p < 2) {
        _Float16* dst = (p == 0) ? Qo : Ko;      // row-major [BS][64]
        #pragma unroll
        for (int nt = 0; nt < 4; ++nt)
            #pragma unroll
            for (int j = 0; j < 8; ++j) {
                const int row = m0 + hid * 8 + j;
                dst[(size_t)row * DOUT + nt * 16 + lr] = (_Float16)(c[nt][j] * scale);
            }
    } else {
        // V transposed: [B][64][S] so P@V B-fragments are contiguous
        const int b  = m0 / SEQ;
        const int s0 = m0 % SEQ;
        #pragma unroll
        for (int nt = 0; nt < 4; ++nt)
            #pragma unroll
            for (int j = 0; j < 8; ++j) {
                Vt[(size_t)(b * DOUT + nt * 16 + lr) * SEQ + s0 + hid * 8 + j] =
                    (_Float16)c[nt][j];
            }
    }
}

// ---------------------------------------------------------------------------
// Kernel 2: flash attention. Block = 4 waves = 64 queries (one batch).
// K/V tiles (32 keys) double-buffered in LDS via async copies.
// ---------------------------------------------------------------------------
__global__ __launch_bounds__(128) void attn_kernel(
        const _Float16* __restrict__ Q, const _Float16* __restrict__ K,
        const _Float16* __restrict__ Vt, float* __restrict__ out)
{
    __shared__ __attribute__((aligned(16))) _Float16 ktile[2][32][DOUT]; // 8 KB
    __shared__ __attribute__((aligned(16))) _Float16 vtile[2][DOUT][32]; // 8 KB
    __shared__ __attribute__((aligned(16))) _Float16 plds[4][16][32];    // 4 KB

    const int tid  = threadIdx.x;
    const int lane = tid & 31;
    const int warp = tid >> 5;
    const int m0   = (blockIdx.x * 4 + warp) * 16;   // flattened query row
    const int b    = (blockIdx.x * 64) / SEQ;        // same for whole block
    const int hid  = lane >> 4;
    const int lr   = lane & 15;

    const _Float16* Kb = K  + (size_t)b * SEQ * DOUT;
    const _Float16* Vb = Vt + (size_t)b * DOUT * SEQ;

    // Q fragments: 2 chunks of K=32 covering D_out=64
    v16h aq[2];
    #pragma unroll
    for (int kc = 0; kc < 2; ++kc) {
        const _Float16* qr = Q + (size_t)(m0 + lr) * DOUT + kc * 32 + hid * 8;
        #pragma unroll
        for (int j = 0; j < 8; ++j) { aq[kc][j] = qr[j]; aq[kc][8 + j] = qr[16 + j]; }
    }

    v8f acc[4] = {v8f{}, v8f{}, v8f{}, v8f{}};
    float m_i[8], l_i[8];
    #pragma unroll
    for (int j = 0; j < 8; ++j) { m_i[j] = -1e30f; l_i[j] = 0.0f; }

    // cooperative async prefetch of one 32-key K/V tile (4 x b128 per wave)
    auto prefetch = [&](int buf, int ks) {
        const _Float16* gk = Kb + (size_t)ks * DOUT;       // contiguous 4 KB
        _Float16*       lk = &ktile[buf][0][0];
        async_copy16(gk + (size_t)tid * 8,         lk + tid * 8);
        async_copy16(gk + (size_t)(tid + 128) * 8, lk + (tid + 128) * 8);
        const _Float16* gv = Vb + ks;                      // 64 rows x 64 B
        _Float16*       lv = &vtile[buf][0][0];
        const int i0 = tid, i1 = tid + 128;                // row=i/4, 16B sub=i%4
        async_copy16(gv + (size_t)(i0 >> 2) * SEQ + (i0 & 3) * 8, lv + i0 * 8);
        async_copy16(gv + (size_t)(i1 >> 2) * SEQ + (i1 & 3) * 8, lv + i1 * 8);
    };

    constexpr int NT = SEQ / 32;   // 128 key tiles
    prefetch(0, 0);

    for (int it = 0; it < NT; ++it) {
        const int buf = it & 1;
        if (it + 1 < NT) { prefetch(buf ^ 1, (it + 1) * 32); async_wait<4>(); }
        else             { async_wait<0>(); }
        __syncthreads();                      // tile[buf] visible to all waves

        // ---- scores: S = Q @ K^T for 32 keys (two 16x16 tiles) ----
        v8f c0 = {}, c1 = {};
        #pragma unroll
        for (int kc = 0; kc < 2; ++kc) {
            const _Float16* kr0 = &ktile[buf][lr][kc * 32 + hid * 8];
            const _Float16* kr1 = &ktile[buf][16 + lr][kc * 32 + hid * 8];
            v16h b0, b1;
            #pragma unroll
            for (int j = 0; j < 8; ++j) {
                b0[j] = kr0[j]; b0[8 + j] = kr0[16 + j];
                b1[j] = kr1[j]; b1[8 + j] = kr1[16 + j];
            }
            c0 = WMMA_F16(aq[kc], b0, c0);
            c1 = WMMA_F16(aq[kc], b1, c1);
        }

        // ---- online softmax (base-2; Q pre-scaled by 1/8 * log2(e)) ----
        float p0[8], p1[8], corr[8];
        #pragma unroll
        for (int j = 0; j < 8; ++j) {
            float s = fmaxf(c0[j], c1[j]);
            #pragma unroll
            for (int mk = 1; mk < 16; mk <<= 1) s = fmaxf(s, __shfl_xor(s, mk, 32));
            const float mnew = fmaxf(m_i[j], s);
            corr[j] = fast_exp2(m_i[j] - mnew);
            p0[j]   = fast_exp2(c0[j] - mnew);
            p1[j]   = fast_exp2(c1[j] - mnew);
            float rs = p0[j] + p1[j];
            #pragma unroll
            for (int mk = 1; mk < 16; mk <<= 1) rs += __shfl_xor(rs, mk, 32);
            l_i[j] = l_i[j] * corr[j] + rs;
            m_i[j] = mnew;
        }
        #pragma unroll
        for (int nt = 0; nt < 4; ++nt)
            #pragma unroll
            for (int j = 0; j < 8; ++j) acc[nt][j] *= corr[j];

        // ---- P: C-layout -> A-layout via per-wave LDS slice ----
        #pragma unroll
        for (int j = 0; j < 8; ++j) {
            const int row = hid * 8 + j;
            plds[warp][row][lr]      = (_Float16)p0[j];
            plds[warp][row][16 + lr] = (_Float16)p1[j];
        }
        v16h ap;
        #pragma unroll
        for (int j = 0; j < 8; ++j) {
            ap[j]     = plds[warp][lr][hid * 8 + j];
            ap[8 + j] = plds[warp][lr][16 + hid * 8 + j];
        }

        // ---- O += P @ V from LDS (vtile rows contiguous along keys) ----
        #pragma unroll
        for (int nt = 0; nt < 4; ++nt) {
            const _Float16* vr = &vtile[buf][nt * 16 + lr][hid * 8];
            v16h vb;
            #pragma unroll
            for (int j = 0; j < 8; ++j) { vb[j] = vr[j]; vb[8 + j] = vr[16 + j]; }
            acc[nt] = WMMA_F16(ap, vb, acc[nt]);
        }
        __syncthreads();                      // readers done before next overwrite
    }

    // ---- epilogue: normalize and store f32 output ----
    float inv[8];
    #pragma unroll
    for (int j = 0; j < 8; ++j) inv[j] = 1.0f / l_i[j];
    #pragma unroll
    for (int nt = 0; nt < 4; ++nt)
        #pragma unroll
        for (int j = 0; j < 8; ++j) {
            const int row = m0 + hid * 8 + j;
            out[(size_t)row * DOUT + nt * 16 + lr] = acc[nt][j] * inv[j];
        }
}

// ---------------------------------------------------------------------------
extern "C" void kernel_launch(void* const* d_in, const int* in_sizes, int n_in,
                              void* d_out, int out_size, void* d_ws, size_t ws_size,
                              hipStream_t stream) {
    const float* x = (const float*)d_in[0];   // [4,4096,128] f32
    const float* w = (const float*)d_in[1];   // [3,128,64]   f32

    _Float16* Q = (_Float16*)d_ws;                 // [BS][64] f16 (pre-scaled)
    _Float16* K = Q + (size_t)BS * DOUT;           // [BS][64] f16
    _Float16* V = K + (size_t)BS * DOUT;           // [B][64][S] f16 (transposed)

    dim3 blk(128);
    proj_kernel<<<dim3((3 * (BS / 16)) / 4), blk, 0, stream>>>(x, w, Q, K, V);
    attn_kernel<<<dim3((BS / 16) / 4), blk, 0, stream>>>(Q, K, V, (float*)d_out);
}
